// FFMLayer_57535381897664
// MI455X (gfx1250) — matrix operand order repaired and checked by reading, compiled, not run on hardware
//
#include <hip/hip_runtime.h>
#include <hip/hip_bf16.h>
#include <math.h>

#define FNUM 26
#define PER_FIELD 40000
#define VTOT (FNUM * PER_FIELD)
#define DIM 16
#define LINN 13
#define MAX_NORM 0.01f

#define NPAIRS 325      // FNUM*(FNUM-1)/2
#define NGROUPS 11      // ceil(325/32)
#define PAIR_PAD 352    // NGROUPS*32
#define ZROW (FNUM*FNUM)

typedef __attribute__((ext_vector_type(16))) __bf16 v16bf;
typedef __attribute__((ext_vector_type(8)))  float  v8f;

struct alignas(16) U4 { unsigned int x, y, z, w; };
union BfVec { v16bf v; U4 q[2]; };

__device__ inline unsigned short f32_to_bf16_rne(float f) {
  unsigned int u = __float_as_uint(f);
  u += 0x7FFFu + ((u >> 16) & 1u);
  return (unsigned short)(u >> 16);
}

__global__ __launch_bounds__(256) void ffm_fused_kernel(
    const float* __restrict__ dense,     // [B,LINN]
    const int*   __restrict__ sparse,    // [B,FNUM]
    const float* __restrict__ tables,    // [FNUM,VTOT,DIM]
    const float* __restrict__ w_sparse,  // [VTOT,1]
    const float* __restrict__ w_dense,   // [LINN]
    const float* __restrict__ b_dense,   // [1]
    float*       __restrict__ out)       // [B,1]
{
  __shared__ __align__(16) unsigned short rows[FNUM*FNUM + 1][DIM]; // bf16 rows + zero row
  __shared__ int   sP[PAIR_PAD];
  __shared__ int   sQ[PAIR_PAD];
  __shared__ int   sIdx[FNUM];
  __shared__ float sPart[8];
  __shared__ float sLinS[FNUM];
  __shared__ float sLinD[LINN];

  const int tid = threadIdx.x;
  const int b   = blockIdx.x;

  // ---- stage 0: indices, pair tables, zero-pad row
  if (tid < FNUM) sIdx[tid] = sparse[b * FNUM + tid] + tid * PER_FIELD;
  for (int p = tid; p < PAIR_PAD; p += 256) {
    if (p < NPAIRS) {
      int i = 0, base = 0;
      while (base + (FNUM - 1 - i) <= p) { base += (FNUM - 1 - i); ++i; }
      int j = i + 1 + (p - base);
      sP[p] = i * FNUM + j;
      sQ[p] = j * FNUM + i;
    } else {
      sP[p] = ZROW;
      sQ[p] = ZROW;
    }
  }
  if (tid == 0) {
    U4 z; z.x = z.y = z.z = z.w = 0u;
    U4* zr = (U4*)(&rows[ZROW][0]);
    zr[0] = z; zr[1] = z;
  }
  __syncthreads();

  // ---- stage 1: gather rows (64B coalesced), renorm, bf16 -> LDS; linear partials
  for (int r = tid; r < FNUM * FNUM; r += 256) {
    int i = r / FNUM, j = r % FNUM;
    if (i == j) continue;  // diagonal rows never used by the i<j pair sum
    const float* src = tables + ((size_t)i * VTOT + (size_t)sIdx[j]) * DIM;
    float4 x0 = ((const float4*)src)[0];
    float4 x1 = ((const float4*)src)[1];
    float4 x2 = ((const float4*)src)[2];
    float4 x3 = ((const float4*)src)[3];
    float f[16] = { x0.x, x0.y, x0.z, x0.w,  x1.x, x1.y, x1.z, x1.w,
                    x2.x, x2.y, x2.z, x2.w,  x3.x, x3.y, x3.z, x3.w };
    float nn = 0.f;
#pragma unroll
    for (int d = 0; d < 16; ++d) nn += f[d] * f[d];
    float norm = sqrtf(nn);
    float sc = (norm > MAX_NORM) ? (MAX_NORM / (norm + 1e-7f)) : 1.0f;
    unsigned int wpk[8];
#pragma unroll
    for (int d = 0; d < 8; ++d) {
      unsigned int lo = f32_to_bf16_rne(f[2 * d]     * sc);
      unsigned int hi = f32_to_bf16_rne(f[2 * d + 1] * sc);
      wpk[d] = lo | (hi << 16);
    }
    U4* dst = (U4*)(&rows[r][0]);
    U4 a, c;
    a.x = wpk[0]; a.y = wpk[1]; a.z = wpk[2]; a.w = wpk[3];
    c.x = wpk[4]; c.y = wpk[5]; c.z = wpk[6]; c.w = wpk[7];
    dst[0] = a; dst[1] = c;
  }
  if (tid >= 64 && tid < 64 + FNUM) {
    int j = tid - 64;
    sLinS[j] = w_sparse[sIdx[j]];
  }
  if (tid >= 96 && tid < 96 + LINN) {
    int l = tid - 96;
    sLinD[l] = dense[b * LINN + l] * w_dense[l];
  }
  __syncthreads();

  // ---- stage 2: pair-dot sum via bf16 WMMA, 32 pairs per instruction
  const int lane = tid & 31;
  const int wv   = tid >> 5;
  const int m    = lane & 15;
  const int h    = lane >> 4;
  const U4* rq = (const U4*)(&rows[0][0]);   // each row = 2 x U4

  v8f acc = {};
  for (int g = wv; g < NGROUPS; g += 8) {    // wave-uniform loop (EXEC all-ones)
    int p0 = g * 32;
    BfVec A, Bm;
    // A[m][k] : k<16 -> P[p0+m][k], k>=16 -> P[p0+16+m][k-16]; lane needs dims 8h..8h+7
    A.q[0] = rq[sP[p0 + m]      * 2 + h];
    A.q[1] = rq[sP[p0 + 16 + m] * 2 + h];
    // B[k][n] : k<16 -> Q[p0+n][k], k>=16 -> Q[p0+16+n][k-16]; lane needs dims 8h..8h+7
    Bm.q[0] = rq[sQ[p0 + m]      * 2 + h];
    Bm.q[1] = rq[sQ[p0 + 16 + m] * 2 + h];
    acc = __builtin_amdgcn_wmma_f32_16x16x32_bf16(
        /*neg_a=*/false, A.v, /*neg_b=*/false, Bm.v,
        /*c_mod=*/(short)0, acc, /*reuse_a=*/false, /*reuse_b=*/false);
  }

  // diagonal of C: lanes 0..7 at vgpr=lane, lanes 24..31 at vgpr=lane-24
  float dsum = 0.0f;
#pragma unroll
  for (int r = 0; r < 8; ++r) {
    dsum = (lane == r)      ? acc[r] : dsum;
    dsum = (lane == 24 + r) ? acc[r] : dsum;
  }
#pragma unroll
  for (int off = 16; off >= 1; off >>= 1)
    dsum += __shfl_xor(dsum, off, 32);
  if (lane == 0) sPart[wv] = dsum;
  __syncthreads();

  // ---- stage 3: deterministic final reduction + linear + sigmoid
  if (tid == 0) {
    float cross = 0.f;
#pragma unroll
    for (int i = 0; i < 8; ++i) cross += sPart[i];
    float lin = b_dense[0];
#pragma unroll
    for (int l = 0; l < LINN; ++l) lin += sLinD[l];
#pragma unroll
    for (int j = 0; j < FNUM; ++j) lin += sLinS[j];
    float z = cross + lin;
    out[b] = 1.0f / (1.0f + expf(-z));
  }
}

extern "C" void kernel_launch(void* const* d_in, const int* in_sizes, int n_in,
                              void* d_out, int out_size, void* d_ws, size_t ws_size,
                              hipStream_t stream) {
  const float* dense    = (const float*)d_in[0];
  const int*   sparse   = (const int*)  d_in[1];
  const float* tables   = (const float*)d_in[2];
  const float* w_sparse = (const float*)d_in[3];
  const float* w_dense  = (const float*)d_in[4];
  const float* b_dense  = (const float*)d_in[5];
  float* out = (float*)d_out;
  const int B = out_size;  // one output per batch element
  ffm_fused_kernel<<<B, 256, 0, stream>>>(dense, sparse, tables, w_sparse,
                                          w_dense, b_dense, out);
}
